// StructureModule_36060545417659
// MI455X (gfx1250) — matrix operand order repaired and checked by reading, not compiled, not run on hardware
//
#include <hip/hip_runtime.h>
#include <hip/hip_bf16.h>
#include <math.h>

// CDNA5 / gfx1250 (MI455X), wave32. WMMA f16->f32 16x16x32 for all GEMMs.
// Workspace requirement: ~22 MB of d_ws (see kernel_launch allocator).

typedef __attribute__((ext_vector_type(16))) _Float16 v16h;
typedef __attribute__((ext_vector_type(8)))  float    v8f;

namespace {
constexpr int   SEQ  = 512;
constexpr int   DIM  = 384;
constexpr int   HEADS = 8;
constexpr int   CATW = 3328;                    // H*(16 + 384 + 4*4)
constexpr float SCALAR_SCALE = 0.14433756729740643f;  // (3*16)^-0.5
constexpr float POINT_SCALE  = 0.13608276348795434f;  // (3*4*4.5)^-0.5
constexpr float PAIR_SCALE   = 0.57735026918962576f;  // 3^-0.5
constexpr float EPSV   = 1e-8f;
constexpr float LN_EPS = 1e-5f;
}

// ---------------------------------------------------------------------------
// Generic WMMA GEMM:  C[b] = (relu?)(A[b](MxK,f32) * B[b](KxN,f32) + bias)
// One wave per 16x(16*NT) output tile. K must be a multiple of 32.
// Bounds handling is via ADDRESS CLAMPS (no predicated loads): clamped A rows
// only pollute accumulator rows >= M (never stored); clamped B cols only
// pollute accumulator cols >= Ncols (never stored). All loads unconditional.
// A lane layout (16-bit A 16x32, ISA 7.12.2): lane l: M=l%16; half p:
//   K = (p<8 ? p : p+8) + (l<16?0:8)  -> contiguous runs {0..7} and {16..23}
//   from (A + row*lda + hi*8), enabling float4 loads.
// B (16-bit B 32x16): lane l: N=l%16; half p: K = (p<8 ? p : p+8) + (l<16?0:8), same order as A
// C/D (f32 16x16): vgpr v: M = v + (l<16?0:8), N = l%16
// ---------------------------------------------------------------------------
template<int NT, bool NTB>
__global__ __launch_bounds__(32)
void gemm_wmma_f16(const float* __restrict__ A, int lda, long sA,
                   const float* __restrict__ B, int ldb, long sB,
                   float* __restrict__ C, int ldc, long sC,
                   int M, int K, int Ncols,
                   const float* __restrict__ bias, int relu)
{
    const int lane = threadIdx.x;
    const int mt = blockIdx.x, nt = blockIdx.y, bz = blockIdx.z;
    A += (long)bz * sA;  B += (long)bz * sB;  C += (long)bz * sC;
    const int r16 = lane & 15;
    const int hi  = lane >> 4;      // 0 for lanes 0-15, 1 for lanes 16-31
    const int arow  = mt * 16 + r16;
    const int arowc = arow < M ? arow : (M - 1);          // clamp (see above)
    const float* abase = A + (long)arowc * lda + hi * 8;

    int bcolc[NT];
#pragma unroll
    for (int t = 0; t < NT; ++t) {
        const int bc = (nt * NT + t) * 16 + r16;
        bcolc[t] = bc < Ncols ? bc : (Ncols - 1);         // clamp
    }

    v8f acc[NT] = {};
    for (int k0 = 0; k0 < K; k0 += 32) {
        // ---- A: 4x float4 (global_load_b128), branchless ----
        const float4 a0 = *(const float4*)(abase + k0);
        const float4 a1 = *(const float4*)(abase + k0 + 4);
        const float4 a2 = *(const float4*)(abase + k0 + 16);
        const float4 a3 = *(const float4*)(abase + k0 + 20);
        v16h a;
        a[0]=(_Float16)a0.x;  a[1]=(_Float16)a0.y;  a[2]=(_Float16)a0.z;  a[3]=(_Float16)a0.w;
        a[4]=(_Float16)a1.x;  a[5]=(_Float16)a1.y;  a[6]=(_Float16)a1.z;  a[7]=(_Float16)a1.w;
        a[8]=(_Float16)a2.x;  a[9]=(_Float16)a2.y;  a[10]=(_Float16)a2.z; a[11]=(_Float16)a2.w;
        a[12]=(_Float16)a3.x; a[13]=(_Float16)a3.y; a[14]=(_Float16)a3.z; a[15]=(_Float16)a3.w;

#pragma unroll
        for (int t = 0; t < NT; ++t) {
            float bf[16];
#pragma unroll
            for (int p = 0; p < 16; ++p) {
                const long idx = (long)(k0 + (p < 8 ? p : p + 8) + hi * 8) * ldb + bcolc[t];
                bf[p] = NTB ? __builtin_nontemporal_load(&B[idx]) : B[idx];
            }
            v16h b;
#pragma unroll
            for (int p = 0; p < 16; ++p) b[p] = (_Float16)bf[p];
            acc[t] = __builtin_amdgcn_wmma_f32_16x16x32_f16(false, a, false, b,
                                                            (short)0, acc[t], false, false);
        }
    }

#pragma unroll
    for (int t = 0; t < NT; ++t) {
        const int bc = (nt * NT + t) * 16 + r16;
        if (bc < Ncols) {
#pragma unroll
            for (int v = 0; v < 8; ++v) {
                const int row = mt * 16 + v + hi * 8;
                if (row < M) {
                    float o = acc[t][v];
                    if (bias) o += bias[bc];
                    if (relu) o = fmaxf(o, 0.0f);
                    C[(long)row * ldc + bc] = o;
                }
            }
        }
    }
}

// ---------------------------------------------------------------------------
// quaternion -> 3x3 rotation (row-major R[r*3+c])
// ---------------------------------------------------------------------------
__global__ void quat_to_rot(const float* __restrict__ q, float* __restrict__ R)
{
    const int n = blockIdx.x * blockDim.x + threadIdx.x;
    if (n >= SEQ) return;
    const float r = q[n*4+0], i = q[n*4+1], j = q[n*4+2], k = q[n*4+3];
    const float two_s = 2.0f / (r*r + i*i + j*j + k*k);
    float* Rn = R + n*9;
    Rn[0] = 1.0f - two_s*(j*j + k*k); Rn[1] = two_s*(i*j - k*r); Rn[2] = two_s*(i*k + j*r);
    Rn[3] = two_s*(i*j + k*r); Rn[4] = 1.0f - two_s*(i*i + k*k); Rn[5] = two_s*(j*k - i*r);
    Rn[6] = two_s*(i*k - j*r); Rn[7] = two_s*(j*k + i*r); Rn[8] = 1.0f - two_s*(i*i + j*j);
}

// ---------------------------------------------------------------------------
// Rotate raw point projections into global frame, compute |q|^2, |k|^2.
// raw layout: p[n*96 + (h*4+d)*3 + c]; rotated: out[((h*SEQ)+n)*12 + d*3 + r]
// forward rot: out_r = sum_c p_c * R[c*3+r] + t_r
// ---------------------------------------------------------------------------
__global__ void rotate_points(const float* __restrict__ pq, const float* __restrict__ pk,
                              const float* __restrict__ pv, const float* __restrict__ R,
                              const float* __restrict__ trans,
                              float* __restrict__ qpr, float* __restrict__ kpr,
                              float* __restrict__ vpr,
                              float* __restrict__ q2, float* __restrict__ k2)
{
    const int tid = blockIdx.x * blockDim.x + threadIdx.x;
    if (tid >= HEADS * SEQ) return;
    const int h = tid / SEQ, n = tid % SEQ;
    const float* Rn = R + n*9;
    const float* tn = trans + n*3;
    float sq = 0.0f, sk = 0.0f;
    for (int d = 0; d < 4; ++d) {
        const int src = n*96 + (h*4 + d)*3;
        const int dst = ((h*SEQ) + n)*12 + d*3;
#pragma unroll
        for (int which = 0; which < 3; ++which) {
            const float* P = (which == 0) ? pq : (which == 1) ? pk : pv;
            float* O = (which == 0) ? qpr : (which == 1) ? kpr : vpr;
            const float p0 = P[src+0], p1 = P[src+1], p2 = P[src+2];
            const float a0 = p0*Rn[0] + p1*Rn[3] + p2*Rn[6] + tn[0];
            const float a1 = p0*Rn[1] + p1*Rn[4] + p2*Rn[7] + tn[1];
            const float a2 = p0*Rn[2] + p1*Rn[5] + p2*Rn[8] + tn[2];
            O[dst+0] = a0; O[dst+1] = a1; O[dst+2] = a2;
            const float s = a0*a0 + a1*a1 + a2*a2;
            if (which == 0) sq += s;
            if (which == 1) sk += s;
        }
    }
    q2[h*SEQ + n] = sq;
    k2[h*SEQ + n] = sk;
}

// ---------------------------------------------------------------------------
// Pair bias: logits[h,i,j] = PAIR_SCALE * (pair[i,j,:] @ Wpb[:,h] + bpb[h])
// One wave per (i,j): 1536 contiguous bytes per wave (coalesced, streamed
// non-temporally), shfl reduce.
// ---------------------------------------------------------------------------
__global__ __launch_bounds__(256)
void pair_bias_kernel(const float* __restrict__ pair, const float* __restrict__ Wpb,
                      const float* __restrict__ bpb, float* __restrict__ logits)
{
    const int wave = threadIdx.x >> 5;
    const int lane = threadIdx.x & 31;
    const long pidx = (long)blockIdx.x * 8 + wave;      // 0 .. SEQ*SEQ-1
    const int i = (int)(pidx >> 9), j = (int)(pidx & 511);
    const float* pr = pair + pidx * DIM;
    float acc[HEADS] = {};
#pragma unroll
    for (int t = 0; t < 12; ++t) {
        const int d = t * 32 + lane;
        const float v = __builtin_nontemporal_load(&pr[d]);
#pragma unroll
        for (int h = 0; h < HEADS; ++h) acc[h] += v * Wpb[d*HEADS + h];
    }
#pragma unroll
    for (int h = 0; h < HEADS; ++h)
#pragma unroll
        for (int s = 16; s > 0; s >>= 1) acc[h] += __shfl_xor(acc[h], s, 32);
    if (lane == 0) {
#pragma unroll
        for (int h = 0; h < HEADS; ++h)
            logits[((long)h*SEQ + i)*SEQ + j] = (acc[h] + bpb[h]) * PAIR_SCALE;
    }
}

// ---------------------------------------------------------------------------
// Logits: fused scalar-QK + point-QK in one WMMA (K = 16 scalar + 12 point,
// zero-padded to 32). Branchless operand build: per-lane pointer select on hi
// (v_cndmask on address) + multiplicative masks for the K-padding lanes.
//   A halves 0..7  : qs[irow, hi*8 + p]        * SCALAR_SCALE
//   A halves 8..11 : qpr[irow, hi*8 + (p-8)]   * cpw
//   A halves 12..15: qpr[irow, hi*8 + (p-8)]   * (hi ? 0 : cpw)
//   B halves mirror A: ks[jcol, hi*8 + p], kpr[jcol, hi*8 + (p-8)], 12..15 * (hi ? 0 : 1)
// Reads of the 4 masked point slots land in allocator padding (safe).
// ---------------------------------------------------------------------------
__global__ __launch_bounds__(32)
void ipa_logits(const float* __restrict__ qs, const float* __restrict__ ks,
                const float* __restrict__ qpr, const float* __restrict__ kpr,
                const float* __restrict__ q2, const float* __restrict__ k2,
                const float* __restrict__ point_weights, float* __restrict__ logits)
{
    const int lane = threadIdx.x;
    const int it = blockIdx.x, jt = blockIdx.y, h = blockIdx.z;
    const int r16 = lane & 15, hi = lane >> 4;
    const float pw  = logf(1.0f + __expf(point_weights[h]));   // softplus
    const float cpw = POINT_SCALE * pw;
    const int irow = it*16 + r16;
    const int jcol = jt*16 + r16;

    const float* aqs = qs  + irow*128 + h*16 + hi*8;                 // 8 floats
    const float* aqp = qpr + ((long)(h*SEQ) + irow)*12 + hi*8;       // 8 floats
    const float* bsc = ks  + jcol*128 + h*16 + hi*8;                 // 8 floats
    const float* bpt = kpr + ((long)(h*SEQ) + jcol)*12 + hi*8;       // 8 floats (masked tail)
    const float* bb  = hi ? bpt : bsc;

    const float4 q0 = *(const float4*)(aqs);
    const float4 q1 = *(const float4*)(aqs + 4);
    const float4 p0 = *(const float4*)(aqp);
    const float4 p1 = *(const float4*)(aqp + 4);
    const float4 b0 = *(const float4*)(bsc);
    const float4 b1 = *(const float4*)(bsc + 4);
    const float4 b2 = *(const float4*)(bpt);
    const float4 b3 = *(const float4*)(bpt + 4);

    const float mhi = hi ? 0.0f : cpw;   // A halves 12..15
    const float bm  = hi ? 0.0f : 1.0f;  // B halves 12..15

    v16h a, b;
    a[0]=(_Float16)(q0.x*SCALAR_SCALE); a[1]=(_Float16)(q0.y*SCALAR_SCALE);
    a[2]=(_Float16)(q0.z*SCALAR_SCALE); a[3]=(_Float16)(q0.w*SCALAR_SCALE);
    a[4]=(_Float16)(q1.x*SCALAR_SCALE); a[5]=(_Float16)(q1.y*SCALAR_SCALE);
    a[6]=(_Float16)(q1.z*SCALAR_SCALE); a[7]=(_Float16)(q1.w*SCALAR_SCALE);
    a[8]=(_Float16)(p0.x*cpw);  a[9]=(_Float16)(p0.y*cpw);
    a[10]=(_Float16)(p0.z*cpw); a[11]=(_Float16)(p0.w*cpw);
    a[12]=(_Float16)(p1.x*mhi); a[13]=(_Float16)(p1.y*mhi);
    a[14]=(_Float16)(p1.z*mhi); a[15]=(_Float16)(p1.w*mhi);

    b[0]=(_Float16)b0.x;  b[1]=(_Float16)b0.y;  b[2]=(_Float16)b0.z;  b[3]=(_Float16)b0.w;
    b[4]=(_Float16)b1.x;  b[5]=(_Float16)b1.y;  b[6]=(_Float16)b1.z;  b[7]=(_Float16)b1.w;
    b[8]=(_Float16)b2.x;  b[9]=(_Float16)b2.y;  b[10]=(_Float16)b2.z; b[11]=(_Float16)b2.w;
    b[12]=(_Float16)(b3.x*bm); b[13]=(_Float16)(b3.y*bm);
    b[14]=(_Float16)(b3.z*bm); b[15]=(_Float16)(b3.w*bm);

    v8f acc = {};
    acc = __builtin_amdgcn_wmma_f32_16x16x32_f16(false, a, false, b,
                                                 (short)0, acc, false, false);
#pragma unroll
    for (int v = 0; v < 8; ++v) {
        const int i = it*16 + v + hi*8;
        const long idx = ((long)h*SEQ + i)*SEQ + jcol;
        const float bias = logits[idx];
        logits[idx] = acc[v] - 0.5f * cpw * (q2[h*SEQ + i] + k2[h*SEQ + jcol]) + bias;
    }
}

// ---------------------------------------------------------------------------
// In-place softmax, one wave per (h,i) row of 512.
// ---------------------------------------------------------------------------
__global__ __launch_bounds__(32)
void softmax_rows(float* __restrict__ logits)
{
    const long row = blockIdx.x;               // h*SEQ + i
    const int lane = threadIdx.x;
    float* p = logits + row * SEQ;
    float vals[16];
    float m = -1e30f;
#pragma unroll
    for (int t = 0; t < 16; ++t) { vals[t] = p[lane + t*32]; m = fmaxf(m, vals[t]); }
#pragma unroll
    for (int s = 16; s > 0; s >>= 1) m = fmaxf(m, __shfl_xor(m, s, 32));
    float sum = 0.0f;
#pragma unroll
    for (int t = 0; t < 16; ++t) { vals[t] = __expf(vals[t] - m); sum += vals[t]; }
#pragma unroll
    for (int s = 16; s > 0; s >>= 1) sum += __shfl_xor(sum, s, 32);
    const float inv = 1.0f / sum;
#pragma unroll
    for (int t = 0; t < 16; ++t) p[lane + t*32] = vals[t] * inv;
}

// ---------------------------------------------------------------------------
// Inverse rotation of attended points + norms, writes into concat buffer.
// inverse: out_r = sum_c (p_c - t_c) * R[r*3+c]
// cat layout per n: [0,128)=res_s, [128,224)=res_p, [224,256)=res_pn, [256,3328)=res_pair
// ---------------------------------------------------------------------------
__global__ void inv_rotate_norm(const float* __restrict__ resp, const float* __restrict__ R,
                                const float* __restrict__ trans, float* __restrict__ cat)
{
    const int tid = blockIdx.x * blockDim.x + threadIdx.x;
    if (tid >= HEADS * SEQ) return;
    const int h = tid / SEQ, n = tid % SEQ;
    const float* Rn = R + n*9;
    const float* tn = trans + n*3;
    for (int d = 0; d < 4; ++d) {
        const int src = ((h*SEQ) + n)*12 + d*3;
        const float p0 = resp[src+0] - tn[0];
        const float p1 = resp[src+1] - tn[1];
        const float p2 = resp[src+2] - tn[2];
        const float l0 = p0*Rn[0] + p1*Rn[1] + p2*Rn[2];
        const float l1 = p0*Rn[3] + p1*Rn[4] + p2*Rn[5];
        const float l2 = p0*Rn[6] + p1*Rn[7] + p2*Rn[8];
        float* cn = cat + (long)n * CATW;
        cn[128 + h*12 + d*3 + 0] = l0;
        cn[128 + h*12 + d*3 + 1] = l1;
        cn[128 + h*12 + d*3 + 2] = l2;
        cn[224 + h*4 + d] = sqrtf(l0*l0 + l1*l1 + l2*l2 + EPSV);
    }
}

// ---------------------------------------------------------------------------
// Residual + LayerNorm, one wave per row of 384.
// ---------------------------------------------------------------------------
__global__ __launch_bounds__(32)
void residual_ln(const float* __restrict__ y, const float* __restrict__ resid,
                 const float* __restrict__ g, const float* __restrict__ beta,
                 float* __restrict__ out)
{
    const int n = blockIdx.x;
    const int lane = threadIdx.x;
    const float* yr = y + (long)n * DIM;
    const float* rr = resid + (long)n * DIM;
    float v[12];
    float s = 0.0f;
#pragma unroll
    for (int t = 0; t < 12; ++t) { v[t] = yr[lane + t*32] + rr[lane + t*32]; s += v[t]; }
#pragma unroll
    for (int k = 16; k > 0; k >>= 1) s += __shfl_xor(s, k, 32);
    const float mean = s * (1.0f / DIM);
    float vs = 0.0f;
#pragma unroll
    for (int t = 0; t < 12; ++t) { const float d = v[t] - mean; vs += d*d; }
#pragma unroll
    for (int k = 16; k > 0; k >>= 1) vs += __shfl_xor(vs, k, 32);
    const float inv = rsqrtf(vs * (1.0f / DIM) + LN_EPS);
#pragma unroll
    for (int t = 0; t < 12; ++t) {
        const int d = lane + t*32;
        out[(long)n*DIM + d] = (v[t] - mean) * inv * g[d] + beta[d];
    }
}

// ---------------------------------------------------------------------------
// Final projection to 6, quaternion multiply, translation update.
// out: [0,1536) translations_new, [1536,3584) quaternions_new
// ---------------------------------------------------------------------------
__global__ void final_update(const float* __restrict__ x2, const float* __restrict__ Wproj,
                             const float* __restrict__ bproj, const float* __restrict__ quats,
                             const float* __restrict__ trans, const float* __restrict__ R,
                             float* __restrict__ out)
{
    const int n = blockIdx.x * blockDim.x + threadIdx.x;
    if (n >= SEQ) return;
    float u[6];
#pragma unroll
    for (int o = 0; o < 6; ++o) u[o] = bproj[o];
    for (int k = 0; k < DIM; ++k) {
        const float xv = x2[(long)n*DIM + k];
#pragma unroll
        for (int o = 0; o < 6; ++o) u[o] += xv * Wproj[k*6 + o];
    }
    const float aw = quats[n*4+0], ax = quats[n*4+1], ay = quats[n*4+2], az = quats[n*4+3];
    const float bx = u[0], by = u[1], bz = u[2];
    out[1536 + n*4 + 0] = aw - ax*bx - ay*by - az*bz;
    out[1536 + n*4 + 1] = aw*bx + ax + ay*bz - az*by;
    out[1536 + n*4 + 2] = aw*by - ax*bz + ay + az*bx;
    out[1536 + n*4 + 3] = aw*bz + ax*by - ay*bx + az;
    const float* Rn = R + n*9;
#pragma unroll
    for (int r = 0; r < 3; ++r)
        out[n*3 + r] = trans[n*3 + r]
                     + u[3]*Rn[0*3 + r] + u[4]*Rn[1*3 + r] + u[5]*Rn[2*3 + r];
}

// ---------------------------------------------------------------------------
extern "C" void kernel_launch(void* const* d_in, const int* in_sizes, int n_in,
                              void* d_out, int out_size, void* d_ws, size_t ws_size,
                              hipStream_t stream)
{
    (void)in_sizes; (void)n_in; (void)out_size; (void)ws_size;
    const float* target = (const float*)d_in[0];
    const float* pair   = (const float*)d_in[1];
    const float* trans  = (const float*)d_in[2];
    const float* quats  = (const float*)d_in[3];
    const float* Wsq    = (const float*)d_in[4];
    const float* Wsk    = (const float*)d_in[5];
    const float* Wsv    = (const float*)d_in[6];
    const float* Wpq    = (const float*)d_in[7];
    const float* Wpk    = (const float*)d_in[8];
    const float* Wpv    = (const float*)d_in[9];
    const float* pwts   = (const float*)d_in[10];
    const float* Wpb    = (const float*)d_in[11];
    const float* bpb    = (const float*)d_in[12];
    const float* Wout   = (const float*)d_in[13];
    const float* bout   = (const float*)d_in[14];
    const float* attn_g = (const float*)d_in[15];
    const float* attn_b = (const float*)d_in[16];
    const float* W1     = (const float*)d_in[17];
    const float* b1     = (const float*)d_in[18];
    const float* W2     = (const float*)d_in[19];
    const float* b2     = (const float*)d_in[20];
    const float* W3     = (const float*)d_in[21];
    const float* b3     = (const float*)d_in[22];
    const float* ff_g   = (const float*)d_in[23];
    const float* ff_b   = (const float*)d_in[24];
    const float* Wproj  = (const float*)d_in[25];
    const float* bproj  = (const float*)d_in[26];
    float* outp = (float*)d_out;

    // ---- workspace carve-out (floats, 256B aligned regions) ----
    float* wsf = (float*)d_ws;
    size_t cur = 0;
    auto alloc = [&](size_t nf) { float* p = wsf + cur; cur += (nf + 63) & ~(size_t)63; return p; };
    float* R      = alloc(SEQ * 9);
    float* qsB    = alloc(SEQ * 128);
    float* ksB    = alloc(SEQ * 128);
    float* vsB    = alloc(SEQ * 128);
    float* pqB    = alloc(SEQ * 96);
    float* pkB    = alloc(SEQ * 96);
    float* pvB    = alloc(SEQ * 96);
    float* qpr    = alloc(HEADS * SEQ * 12 + 16);   // +pad: masked K-lane reads
    float* kpr    = alloc(HEADS * SEQ * 12 + 16);
    float* vpr    = alloc(HEADS * SEQ * 12 + 16);
    float* q2B    = alloc(HEADS * SEQ);
    float* k2B    = alloc(HEADS * SEQ);
    float* logits = alloc((size_t)HEADS * SEQ * SEQ);   // bias -> logits -> attn (in place)
    float* respB  = alloc(HEADS * SEQ * 12 + 16);
    float* catB   = alloc((size_t)SEQ * CATW);
    float* tmpB   = alloc(SEQ * DIM);
    float* x1B    = alloc(SEQ * DIM);
    float* y1B    = alloc(SEQ * DIM);
    float* y2B    = alloc(SEQ * DIM);
    float* y3B    = alloc(SEQ * DIM);

    // 1) rotations
    quat_to_rot<<<8, 64, 0, stream>>>(quats, R);

    // 2) projections (WMMA GEMMs, 2 col-tiles per wave): x(512x384) @ W
    gemm_wmma_f16<2,false><<<dim3(32, 4, 1), 32, 0, stream>>>(
        target, DIM, 0, Wsq, 128, 0, qsB, 128, 0, SEQ, DIM, 128, nullptr, 0);
    gemm_wmma_f16<2,false><<<dim3(32, 4, 1), 32, 0, stream>>>(
        target, DIM, 0, Wsk, 128, 0, ksB, 128, 0, SEQ, DIM, 128, nullptr, 0);
    gemm_wmma_f16<2,false><<<dim3(32, 4, 1), 32, 0, stream>>>(
        target, DIM, 0, Wsv, 128, 0, vsB, 128, 0, SEQ, DIM, 128, nullptr, 0);
    gemm_wmma_f16<2,false><<<dim3(32, 3, 1), 32, 0, stream>>>(
        target, DIM, 0, Wpq,  96, 0, pqB,  96, 0, SEQ, DIM,  96, nullptr, 0);
    gemm_wmma_f16<2,false><<<dim3(32, 3, 1), 32, 0, stream>>>(
        target, DIM, 0, Wpk,  96, 0, pkB,  96, 0, SEQ, DIM,  96, nullptr, 0);
    gemm_wmma_f16<2,false><<<dim3(32, 3, 1), 32, 0, stream>>>(
        target, DIM, 0, Wpv,  96, 0, pvB,  96, 0, SEQ, DIM,  96, nullptr, 0);

    // 3) rotate points into global frame, |q|^2 / |k|^2
    rotate_points<<<32, 128, 0, stream>>>(pqB, pkB, pvB, R, trans, qpr, kpr, vpr, q2B, k2B);

    // 4) pair bias (streams 402 MB of pair_repr once, non-temporal)
    pair_bias_kernel<<<(SEQ * SEQ) / 8, 256, 0, stream>>>(pair, Wpb, bpb, logits);

    // 5) fused scalar+point QK logits (WMMA), adds bias + distance terms in place
    ipa_logits<<<dim3(32, 32, 8), 32, 0, stream>>>(qsB, ksB, qpr, kpr, q2B, k2B, pwts, logits);

    // 6) softmax (in place -> attn)
    softmax_rows<<<HEADS * SEQ, 32, 0, stream>>>(logits);

    // 7a) res_s = attn @ vs   -> cat[:, h*16 + d]   (batched over heads)
    gemm_wmma_f16<1,false><<<dim3(32, 1, HEADS), 32, 0, stream>>>(
        logits, SEQ, (long)SEQ * SEQ, vsB, 128, 16,
        catB, CATW, 16, SEQ, SEQ, 16, nullptr, 0);
    // 7b) res_p(global) = attn @ vp -> respB[h][n][12]
    gemm_wmma_f16<1,false><<<dim3(32, 1, HEADS), 32, 0, stream>>>(
        logits, SEQ, (long)SEQ * SEQ, vpr, 12, (long)SEQ * 12,
        respB, 12, (long)SEQ * 12, SEQ, SEQ, 12, nullptr, 0);
    // 7c) res_pair[i] = attn[:, i, :](8x512) @ pair[i](512x384) -> cat[:, 256 + h*384 + d]
    //     batched over i; second 402 MB pair stream (non-temporal B loads).
    gemm_wmma_f16<2,true><<<dim3(1, 12, SEQ), 32, 0, stream>>>(
        logits, SEQ * SEQ, (long)SEQ, pair, DIM, (long)SEQ * DIM,
        catB + 256, DIM, (long)CATW, HEADS, SEQ, DIM, nullptr, 0);

    // 8) inverse-rotate attended points + norms into cat
    inv_rotate_norm<<<32, 128, 0, stream>>>(respB, R, trans, catB);

    // 9) attn_out = cat(512x3328) @ Wout + bout ; residual + LN
    gemm_wmma_f16<2,false><<<dim3(32, 12, 1), 32, 0, stream>>>(
        catB, CATW, 0, Wout, DIM, 0, tmpB, DIM, 0, SEQ, CATW, DIM, bout, 0);
    residual_ln<<<SEQ, 32, 0, stream>>>(tmpB, target, attn_g, attn_b, x1B);

    // 10) feed-forward (WMMA GEMMs) + LN
    gemm_wmma_f16<2,false><<<dim3(32, 12, 1), 32, 0, stream>>>(
        x1B, DIM, 0, W1, DIM, 0, y1B, DIM, 0, SEQ, DIM, DIM, b1, 1);
    gemm_wmma_f16<2,false><<<dim3(32, 12, 1), 32, 0, stream>>>(
        y1B, DIM, 0, W2, DIM, 0, y2B, DIM, 0, SEQ, DIM, DIM, b2, 1);
    gemm_wmma_f16<2,false><<<dim3(32, 12, 1), 32, 0, stream>>>(
        y2B, DIM, 0, W3, DIM, 0, y3B, DIM, 0, SEQ, DIM, DIM, b3, 0);
    residual_ln<<<SEQ, 32, 0, stream>>>(y3B, x1B, ff_g, ff_b, x1B /* reuse as x2 */);

    // 11) final frame updates
    final_update<<<4, 128, 0, stream>>>(x1B, Wproj, bproj, quats, trans, R, outp);
}
